// MultiHeadAttentionLayer_14877766714149
// MI455X (gfx1250) — matrix operand order repaired and checked by reading, not compiled, not run on hardware
//
#include <hip/hip_runtime.h>

// ---------------------------------------------------------------------------
// MultiHeadAttentionLayer on MI455X (gfx1250), wave32 + WMMA bf16.
// Pipeline: wconv(W->Wt bf16) x4 ; GEMM Q,K(->K^T),V ; flash-attention ; GEMM O.
// ASYNCcnt-tracked global->LDS staging (GLOBAL_LOAD_ASYNC_TO_LDS_B128) for all
// bf16 tiles, guarded by __has_builtin with a synchronous fallback.
// ---------------------------------------------------------------------------

typedef __bf16 bf16;
typedef __attribute__((ext_vector_type(16))) __bf16        v16bf;
typedef __attribute__((ext_vector_type(8)))  float         v8f;
typedef __attribute__((ext_vector_type(4)))  unsigned int  u32x4;
typedef __attribute__((ext_vector_type(4)))  int           i32x4;
typedef __attribute__((ext_vector_type(4)))  float         f32x4;

union FragU { u32x4 u[2]; v16bf v; };
union BF8   { bf16 h[8];  u32x4 u; };

__device__ __forceinline__ v8f wmma_bf16(v16bf a, v16bf b, v8f c) {
  // (neg_a, A, neg_b, B, c_mod, C, reuse_a, reuse_b)
  return __builtin_amdgcn_wmma_f32_16x16x32_bf16(false, a, false, b, (short)0, c,
                                                 false, false);
}

// wave32 xor-shuffle via ds_swizzle (group-of-32: and=0x1f, xor=XM). xor<=8 stays
// inside each 16-lane half, matching the C-matrix row distribution.
#define SWZ_XOR_F(v, XM)                                                       \
  __builtin_bit_cast(float, __builtin_amdgcn_ds_swizzle(                       \
                                __builtin_bit_cast(int, (v)),                  \
                                (((XM) << 10) | 0x1f)))

// --------------------------- async global->LDS -----------------------------
#if defined(__AMDGCN__) &&                                                     \
    __has_builtin(__builtin_amdgcn_global_load_async_to_lds_b128) &&           \
    __has_builtin(__builtin_amdgcn_s_wait_asynccnt)
#define USE_ASYNC_LDS 1
#else
#define USE_ASYNC_LDS 0
#endif

#if USE_ASYNC_LDS
// Builtin signature (from the compiler diagnostic): arg0 is
// 'int __attribute__((vector_size(16))) __device__ *' i.e. int4 in AS(1);
// arg1 is the LDS destination (int4 in AS(3)); then imm offset, imm cpol.
typedef __attribute__((address_space(1))) i32x4 g1_i32x4;
typedef __attribute__((address_space(3))) i32x4 l3_i32x4;
// Generic->AS3 via low-32-bit truncation (generic LDS VA = aperture | offset);
// generic->AS1 keeps the 64-bit VA. Integer round-trips sidestep the
// "mismatched address space" C++ cast restriction.
__device__ __forceinline__ void async_copy_b128(const bf16* g, bf16* l) {
  __builtin_amdgcn_global_load_async_to_lds_b128(
      (g1_i32x4*)(unsigned long long)(size_t)g,
      (l3_i32x4*)(unsigned int)(size_t)l, 0, 0);
}
__device__ __forceinline__ void async_wait0() {
  __builtin_amdgcn_s_wait_asynccnt(0);
}
#endif

__device__ __forceinline__ void sync_copy_b128(const bf16* g, bf16* l) {
  *(u32x4*)l = *(const u32x4*)g;
}

__device__ __forceinline__ void tile_copy_b128(const bf16* g, bf16* l) {
#if USE_ASYNC_LDS
  async_copy_b128(g, l);
#else
  sync_copy_b128(g, l);
#endif
}

__device__ __forceinline__ void tile_copy_fence() {
#if USE_ASYNC_LDS
  async_wait0();
#endif
}

constexpr int Nb = 4, Sq = 2048, Tk = 2048, Dm = 1024, Hh = 16, DHd = 64;
constexpr int Mrows = Nb * Sq;  // 8192

// ---------------------------------------------------------------------------
// Weight convert + transpose: Wt[d][e] = bf16(W[e][d]), 1024x1024.
// ---------------------------------------------------------------------------
__global__ __launch_bounds__(256) void wconv(const float* __restrict__ W,
                                             bf16* __restrict__ Wt) {
  __shared__ float tile[32][33];
  const int bx = blockIdx.x * 32;  // d
  const int by = blockIdx.y * 32;  // e
  const int tx = threadIdx.x, ty = threadIdx.y;
#pragma unroll
  for (int j = 0; j < 4; ++j) {
    const int e = by + ty + j * 8;
    tile[ty + j * 8][tx] = W[(size_t)e * Dm + bx + tx];
  }
  __syncthreads();
#pragma unroll
  for (int j = 0; j < 4; ++j) {
    const int d = bx + ty + j * 8;
    Wt[(size_t)d * Dm + by + tx] = (bf16)tile[tx][ty + j * 8];
  }
}

// ---------------------------------------------------------------------------
// Tiled WMMA GEMM: Out[m][e] = sum_d X[m][d] * Wt[d][e] + bias[e]
//   BM=128, BN=128, BK=32 ; 256 threads = 8 waves in a 4x2 grid,
//   each wave computes a 32x64 patch = 2x4 tiles of 16x16.
// XBF16: X is bf16 (else f32, converted during the LDS stage).
// MODE 0: bf16 out, row-major [m][e]
// MODE 1: bf16 out, K^T layout [n][h][d][t]   (for attention scores B-operand)
// MODE 2: f32 out, row-major (final projection)
// ---------------------------------------------------------------------------
template <bool XBF16, int MODE>
__global__ __launch_bounds__(256) void gemm_wmma(const void* __restrict__ Xv,
                                                 const bf16* __restrict__ Wt,
                                                 const float* __restrict__ bias,
                                                 void* __restrict__ Outv) {
  constexpr int Kd = Dm, Nd = Dm;
  __shared__ bf16 As[128][32];
  __shared__ bf16 Bs[32][128];

  const int tid = threadIdx.x;
  const int lane = tid & 31, w = tid >> 5;
  const int wr = w >> 1, wc = w & 1;      // wave 4x2 grid
  const int lm = lane & 15;
  const int kb = (lane >> 4) << 3;        // 0 or 8 (upper-half lanes)
  const int m0 = blockIdx.y * 128, n0 = blockIdx.x * 128;

  v8f acc[2][4];
  const v8f zero8 = {0, 0, 0, 0, 0, 0, 0, 0};
#pragma unroll
  for (int mt = 0; mt < 2; ++mt)
#pragma unroll
    for (int nt = 0; nt < 4; ++nt) acc[mt][nt] = zero8;

  const int arow = tid >> 1, aseg = (tid & 1) * 16;   // A: 16 elems / thread
  const int brow = tid >> 3, bseg = (tid & 7) * 16;   // B: 16 elems / thread

  for (int kk = 0; kk < Kd; kk += 32) {
    __syncthreads();
    if constexpr (XBF16) {
      const bf16* src = (const bf16*)Xv + (size_t)(m0 + arow) * Kd + kk + aseg;
      tile_copy_b128(src,     &As[arow][aseg]);
      tile_copy_b128(src + 8, &As[arow][aseg + 8]);
    } else {
      const float* src = (const float*)Xv + (size_t)(m0 + arow) * Kd + kk + aseg;
      f32x4 f0 = *(const f32x4*)(src);
      f32x4 f1 = *(const f32x4*)(src + 4);
      f32x4 f2 = *(const f32x4*)(src + 8);
      f32x4 f3 = *(const f32x4*)(src + 12);
      BF8 p0, p1;
#pragma unroll
      for (int i = 0; i < 4; ++i) {
        p0.h[i] = (bf16)f0[i]; p0.h[4 + i] = (bf16)f1[i];
        p1.h[i] = (bf16)f2[i]; p1.h[4 + i] = (bf16)f3[i];
      }
      *(u32x4*)&As[arow][aseg]     = p0.u;
      *(u32x4*)&As[arow][aseg + 8] = p1.u;
    }
    {
      const bf16* src = Wt + (size_t)(kk + brow) * Nd + n0 + bseg;
      tile_copy_b128(src,     &Bs[brow][bseg]);
      tile_copy_b128(src + 8, &Bs[brow][bseg + 8]);
    }
    tile_copy_fence();
    __syncthreads();

    FragU a[2];
#pragma unroll
    for (int mt = 0; mt < 2; ++mt) {
      const bf16* ap = &As[wr * 32 + mt * 16 + lm][0];
      a[mt].u[0] = *(const u32x4*)(ap + kb);        // K = kb..kb+7
      a[mt].u[1] = *(const u32x4*)(ap + 16 + kb);   // K = 16+kb..
    }
#pragma unroll
    for (int nt = 0; nt < 4; ++nt) {
      FragU b;
#pragma unroll
      for (int s = 0; s < 16; ++s)  // B col n = lm, K in table order
        b.v[s] = Bs[s < 8 ? kb + s : 8 + kb + s][wc * 64 + nt * 16 + lm];
#pragma unroll
      for (int mt = 0; mt < 2; ++mt)
        acc[mt][nt] = wmma_bf16(a[mt].v, b.v, acc[mt][nt]);
    }
  }

  // Epilogue. C layout: VGPR r -> row (r + kb), lane&15 -> col.
#pragma unroll
  for (int mt = 0; mt < 2; ++mt) {
    const int rowbase = m0 + wr * 32 + mt * 16 + kb;
#pragma unroll
    for (int nt = 0; nt < 4; ++nt) {
      const int gcol = n0 + wc * 64 + nt * 16 + lm;
      const float bv = bias[gcol];
      if constexpr (MODE == 0) {
        bf16* out = (bf16*)Outv;
#pragma unroll
        for (int r = 0; r < 8; ++r)
          out[(size_t)(rowbase + r) * Nd + gcol] = (bf16)(acc[mt][nt][r] + bv);
      } else if constexpr (MODE == 2) {
        float* out = (float*)Outv;
#pragma unroll
        for (int r = 0; r < 8; ++r)
          out[(size_t)(rowbase + r) * Nd + gcol] = acc[mt][nt][r] + bv;
      } else {  // MODE == 1: K^T layout [n][h][d][t]; 8 contiguous t per lane
        bf16* out = (bf16*)Outv;
        const int nbat = rowbase >> 11;   // / Tk
        const int t = rowbase & (Tk - 1);
        const int hh = gcol >> 6, d = gcol & 63;
        BF8 pk;
#pragma unroll
        for (int r = 0; r < 8; ++r) pk.h[r] = (bf16)(acc[mt][nt][r] + bv);
        *(u32x4*)(out + ((size_t)((nbat * Hh + hh) * DHd + d)) * Tk + t) = pk.u;
      }
    }
  }
}

// ---------------------------------------------------------------------------
// Flash attention: one (n,h) x 128-query block per CTA, 8 waves x 16 rows.
// Q frags resident; per 128-key block stage K^T(64x128) and V(128x64) in LDS
// (async global->LDS when available), WMMA scores -> online softmax
// (ds_swizzle row reductions) -> P via LDS re-layout -> WMMA P*V accumulation.
// y written as bf16 [n][s][e].
// ---------------------------------------------------------------------------
__global__ __launch_bounds__(256) void flash_attn(const bf16* __restrict__ qb,
                                                  const bf16* __restrict__ ktb,
                                                  const bf16* __restrict__ vb,
                                                  bf16* __restrict__ yb) {
  __shared__ bf16 kt_s[64][128];      // 16 KB
  __shared__ bf16 v_s[128][64];       // 16 KB
  __shared__ bf16 p_s[8][16][128];    // 32 KB, wave-private slabs

  const int tid = threadIdx.x;
  const int lane = tid & 31, w = tid >> 5;
  const int lm = lane & 15;
  const int kb = (lane >> 4) << 3;
  const int nh = blockIdx.y;
  const int n = nh >> 4, h = nh & 15;
  const int s0 = blockIdx.x * 128 + w * 16;

  // Q fragments: rows s0..s0+15, K-dim = DH = 64 as two 16x32 A-frags.
  FragU qf[2];
  {
    const bf16* qp = qb + (size_t)(n * Sq + s0 + lm) * Dm + h * DHd;
#pragma unroll
    for (int c = 0; c < 2; ++c) {
      qf[c].u[0] = *(const u32x4*)(qp + c * 32 + kb);
      qf[c].u[1] = *(const u32x4*)(qp + c * 32 + 16 + kb);
    }
  }

  const v8f zero8 = {0, 0, 0, 0, 0, 0, 0, 0};
  v8f o[4];
#pragma unroll
  for (int jd = 0; jd < 4; ++jd) o[jd] = zero8;
  float m_i[8], l_i[8];
#pragma unroll
  for (int r = 0; r < 8; ++r) { m_i[r] = -3.0e38f; l_i[r] = 0.0f; }

  for (int t0 = 0; t0 < Tk; t0 += 128) {
    __syncthreads();
    {  // K^T tile: rows d=0..63, cols t0..t0+127 (row-major in ktb)
      const int row = tid >> 2, seg = (tid & 3) * 32;
      const bf16* src = ktb + (size_t)((n * Hh + h) * DHd + row) * Tk + t0 + seg;
#pragma unroll
      for (int i = 0; i < 4; ++i)
        tile_copy_b128(src + i * 8, &kt_s[row][seg + i * 8]);
    }
    {  // V tile: rows t, cols d=0..63
      const int row = tid >> 1, seg = (tid & 1) * 32;
      const bf16* src = vb + (size_t)(n * Tk + t0 + row) * Dm + h * DHd + seg;
#pragma unroll
      for (int i = 0; i < 4; ++i)
        tile_copy_b128(src + i * 8, &v_s[row][seg + i * 8]);
    }
    tile_copy_fence();
    __syncthreads();

    // Scores: this wave's 16x128 strip, 8 tiles of 16x16, K = 64 in 2 chunks.
    v8f cs[8];
#pragma unroll
    for (int j = 0; j < 8; ++j) {
      v8f c = zero8;
      FragU b0, b1;
#pragma unroll
      for (int s = 0; s < 16; ++s)  // B col t = lm, d in table order
        b0.v[s] = kt_s[s < 8 ? kb + s : 8 + kb + s][j * 16 + lm];
      c = wmma_bf16(qf[0].v, b0.v, c);
#pragma unroll
      for (int s = 0; s < 16; ++s)  // B col t = lm, d = 32 + table order
        b1.v[s] = kt_s[32 + (s < 8 ? kb + s : 8 + kb + s)][j * 16 + lm];
      c = wmma_bf16(qf[1].v, b1.v, c);
#pragma unroll
      for (int r = 0; r < 8; ++r) c[r] *= 0.125f;   // 1/sqrt(DH)
      cs[j] = c;
    }

    // Online softmax; row r lives in VGPR r of each frag within a lane-half.
    float corr[8];
#pragma unroll
    for (int r = 0; r < 8; ++r) {
      float mx = cs[0][r];
#pragma unroll
      for (int j = 1; j < 8; ++j) mx = fmaxf(mx, cs[j][r]);
      mx = fmaxf(mx, SWZ_XOR_F(mx, 1));
      mx = fmaxf(mx, SWZ_XOR_F(mx, 2));
      mx = fmaxf(mx, SWZ_XOR_F(mx, 4));
      mx = fmaxf(mx, SWZ_XOR_F(mx, 8));
      const float mn = fmaxf(m_i[r], mx);
      corr[r] = __expf(m_i[r] - mn);
      m_i[r] = mn;
    }
    float rs[8];
#pragma unroll
    for (int r = 0; r < 8; ++r) rs[r] = 0.0f;
#pragma unroll
    for (int j = 0; j < 8; ++j)
#pragma unroll
      for (int r = 0; r < 8; ++r) {
        const float p = __expf(cs[j][r] - m_i[r]);
        cs[j][r] = p;
        rs[r] += p;
      }
#pragma unroll
    for (int r = 0; r < 8; ++r) {
      rs[r] += SWZ_XOR_F(rs[r], 1);
      rs[r] += SWZ_XOR_F(rs[r], 2);
      rs[r] += SWZ_XOR_F(rs[r], 4);
      rs[r] += SWZ_XOR_F(rs[r], 8);
      l_i[r] = l_i[r] * corr[r] + rs[r];
    }
#pragma unroll
    for (int jd = 0; jd < 4; ++jd)
#pragma unroll
      for (int r = 0; r < 8; ++r) o[jd][r] *= corr[r];

    // Re-layout P: C-frag -> LDS (wave-private) -> A-frags (DS unit in-order).
#pragma unroll
    for (int j = 0; j < 8; ++j)
#pragma unroll
      for (int r = 0; r < 8; ++r)
        p_s[w][kb + r][j * 16 + lm] = (bf16)cs[j][r];

    // O += P x V : 4 K-chunks of 32 keys, 4 column tiles of DH.
#pragma unroll
    for (int tc = 0; tc < 4; ++tc) {
      FragU pf;
      const bf16* pp = &p_s[w][lm][tc * 32];
      pf.u[0] = *(const u32x4*)(pp + kb);
      pf.u[1] = *(const u32x4*)(pp + 16 + kb);
#pragma unroll
      for (int jd = 0; jd < 4; ++jd) {
        FragU vf;
#pragma unroll
        for (int s = 0; s < 16; ++s)  // B col d = lm, t in table order
          vf.v[s] = v_s[tc * 32 + (s < 8 ? kb + s : 8 + kb + s)][jd * 16 + lm];
        o[jd] = wmma_bf16(pf.v, vf.v, o[jd]);
      }
    }
  }

  // Normalize and store y as bf16 [n][s][h*DH + d].
  float inv[8];
#pragma unroll
  for (int r = 0; r < 8; ++r) inv[r] = 1.0f / l_i[r];
#pragma unroll
  for (int jd = 0; jd < 4; ++jd) {
    const int gcol = h * DHd + jd * 16 + lm;
#pragma unroll
    for (int r = 0; r < 8; ++r) {
      const int srow = s0 + kb + r;
      yb[(size_t)(n * Sq + srow) * Dm + gcol] = (bf16)(o[jd][r] * inv[r]);
    }
  }
}

// ---------------------------------------------------------------------------
// Host orchestration.
// ---------------------------------------------------------------------------
extern "C" void kernel_launch(void* const* d_in, const int* in_sizes, int n_in,
                              void* d_out, int out_size, void* d_ws,
                              size_t ws_size, hipStream_t stream) {
  const float* query = (const float*)d_in[0];
  const float* key_  = (const float*)d_in[1];
  const float* value = (const float*)d_in[2];
  const float* Wq = (const float*)d_in[3];
  const float* bq = (const float*)d_in[4];
  const float* Wk = (const float*)d_in[5];
  const float* bk = (const float*)d_in[6];
  const float* Wv = (const float*)d_in[7];
  const float* bv = (const float*)d_in[8];
  const float* Wo = (const float*)d_in[9];
  const float* bo = (const float*)d_in[10];

  char* ws = (char*)d_ws;
  const size_t wBytes = (size_t)Dm * Dm * sizeof(bf16);        // 2 MB
  const size_t xBytes = (size_t)Nb * Sq * Dm * sizeof(bf16);   // 16 MB
  bf16* Wqt = (bf16*)ws;            ws += wBytes;
  bf16* Wkt = (bf16*)ws;            ws += wBytes;
  bf16* Wvt = (bf16*)ws;            ws += wBytes;
  bf16* Wot = (bf16*)ws;            ws += wBytes;
  bf16* qb  = (bf16*)ws;            ws += xBytes;
  bf16* ktb = (bf16*)ws;            ws += xBytes;   // [n][h][d][t]
  bf16* vb  = (bf16*)ws;            ws += xBytes;
  bf16* yb  = (bf16*)ws;            ws += xBytes;

  const dim3 tb(32, 8), tg(32, 32);
  wconv<<<tg, tb, 0, stream>>>(Wq, Wqt);
  wconv<<<tg, tb, 0, stream>>>(Wk, Wkt);
  wconv<<<tg, tb, 0, stream>>>(Wv, Wvt);
  wconv<<<tg, tb, 0, stream>>>(Wo, Wot);

  const dim3 gg(Dm / 128, Mrows / 128);  // (8, 64)
  gemm_wmma<false, 0><<<gg, 256, 0, stream>>>(query, Wqt, bq, qb);
  gemm_wmma<false, 1><<<gg, 256, 0, stream>>>(key_,  Wkt, bk, ktb);
  gemm_wmma<false, 0><<<gg, 256, 0, stream>>>(value, Wvt, bv, vb);

  flash_attn<<<dim3(Sq / 128, Nb * Hh), 256, 0, stream>>>(qb, ktb, vb, yb);

  gemm_wmma<true, 2><<<gg, 256, 0, stream>>>(yb, Wot, bo, (float*)d_out);
}
